// FeedBack_46703474376732
// MI455X (gfx1250) — compile-verified, matrix operand order checked by reading, not run on hardware
//
#include <hip/hip_runtime.h>
#include <cstring>

// ---------------------------------------------------------------------------
// FeedBack LSTM rollout for MI455X (gfx1250, wave32, WMMA).
// B=2048, S=64, F=3, U=512, 4U=2048, OUT_NOTES=64 (127 total LSTM steps).
//
// 128 persistent workgroups (512 threads = 16 waves), each owning 16 batch
// rows for all 127 steps.  Recurrent GEMM h[16,512] @ W[512,2048] via
// v_wmma_f32_16x16x32_bf16 with pre-swizzled B fragments -> coalesced
// global_load_b128 from L2-resident weights.  Each wave owns 2 unit-tiles x
// 4 gates (8 accumulators = 64 VGPRs) to stay well under 256 VGPRs and keep
// multiple waves per SIMD in flight.  Cell state c in registers, h in LDS.
// Gate nonlinearities are branch-free (exp/rcp only).
// ---------------------------------------------------------------------------

typedef __attribute__((ext_vector_type(16))) __bf16 v16bf;
typedef __attribute__((ext_vector_type(8)))  float  v8f;

#define BATCH   2048
#define SEQ     64
#define FEAT    3
#define UNITS   512
#define GATES4  2048          // 4*UNITS
#define NSTEPS  127           // 64 warmup + 63 autoregressive
#define MTILE   16            // batch rows per workgroup
#define NTHR    512           // threads per workgroup (16 waves)
#define HP      516           // padded h row stride (floats) to spread LDS banks
#define NKF     16            // K fragments of 32 (512/32)

#if defined(__has_builtin)
#if __has_builtin(__builtin_amdgcn_cvt_pk_bf16_f32)
#define HAVE_PK_BF16 1
#endif
#endif

static __device__ __forceinline__ __bf16 f2bf(float x) {
  unsigned u = __builtin_bit_cast(unsigned, x);
  u += 0x7FFFu + ((u >> 16) & 1u);          // round to nearest even
  unsigned short h = (unsigned short)(u >> 16);
  return __builtin_bit_cast(__bf16, h);
}
static __device__ __forceinline__ float bf2f_u16(unsigned short b) {
  unsigned u = ((unsigned)b) << 16;
  return __builtin_bit_cast(float, u);
}
// pack two f32 into one dword of two bf16 (RNE)
static __device__ __forceinline__ unsigned pk_bf16(float x, float y) {
#ifdef HAVE_PK_BF16
  auto pk = __builtin_amdgcn_cvt_pk_bf16_f32(x, y);
  return __builtin_bit_cast(unsigned, pk);
#else
  unsigned ux = __builtin_bit_cast(unsigned, x);
  unsigned uy = __builtin_bit_cast(unsigned, y);
  ux += 0x7FFFu + ((ux >> 16) & 1u);
  uy += 0x7FFFu + ((uy >> 16) & 1u);
  return (ux >> 16) | (uy & 0xFFFF0000u);
#endif
}
// branch-free sigmoid / tanh (v_exp_f32 + v_rcp_f32, saturates correctly)
static __device__ __forceinline__ float sigmoidf_(float z) {
  return __frcp_rn(1.0f + __expf(-z));
}
static __device__ __forceinline__ float tanhf_(float z) {
  // tanh(z) = 2*sigmoid(2z) - 1; exp(+inf)->inf->rcp->0->-1, exp(-inf)->0->+1
  return __builtin_fmaf(2.0f, __frcp_rn(1.0f + __expf(-2.0f * z)), -1.0f);
}

struct AW { unsigned w[8]; };   // 8 dwords == one v16bf fragment

// ---------------------------------------------------------------------------
// Pack rec_kernel [512,2048] f32 -> bf16 WMMA-B fragments:
//   Wp[nt][kf][lane][e].  B tile is K(32) x N(16); lanes 0-15 -> N=lane,
//   K pairs {0,1..6,7} in VGPR0-3 and {16..23} in VGPR4-7; lanes 16-31 ->
//   N=lane-16, K {8..15} / {24..31}  (mirror of documented A layout).
// ---------------------------------------------------------------------------
__global__ void feedback_pack_w(const float* __restrict__ rec,
                                __bf16* __restrict__ Wp) {
  int p = blockIdx.x * blockDim.x + threadIdx.x;   // element index
  if (p >= UNITS * GATES4) return;
  int e    = p & 15;
  int lane = (p >> 4) & 31;
  int kf   = (p >> 9) & 15;
  int nt   = p >> 13;
  int v    = e >> 1;
  int half = e & 1;
  int n    = nt * 16 + (lane & 15);
  int klo  = ((v & 3) * 2) + half + ((lane >= 16) ? 8 : 0) + ((v >= 4) ? 16 : 0);
  int k    = kf * 32 + klo;
  Wp[p] = f2bf(rec[k * GATES4 + n]);
}

// ---------------------------------------------------------------------------
// Main persistent kernel: one WG = 16 batch rows, 16 waves, 127 steps.
// ---------------------------------------------------------------------------
__global__ void __launch_bounds__(NTHR)
feedback_lstm_main(const float* __restrict__ inputs,   // [B,S,F]
                   const float* __restrict__ kern,     // [F,4U]
                   const float* __restrict__ bias,     // [4U]
                   const float* __restrict__ dw,       // [U,F]
                   const float* __restrict__ db,       // [F]
                   const v16bf* __restrict__ Wp,       // packed rec weights
                   float*       __restrict__ out) {    // [64,B,F]
  __shared__ alignas(16) float          h_s[MTILE * HP];      // h state, f32
  __shared__ alignas(16) unsigned short kb_s[GATES4 * 4];     // [n][bias,k0,k1,k2] bf16
  __shared__ alignas(16) float          dw_s[UNITS * FEAT + 4];
  __shared__ alignas(16) float          x_s[MTILE * 4];       // current cell input
  __shared__ alignas(16) float          red_s[MTILE * 32 * 3]; // dense partials

  const int tix  = threadIdx.x;
  const int lane = tix & 31;
  const int wid  = tix >> 5;           // 0..15
  const int b0   = blockIdx.x * MTILE;
  const int Ml   = lane & 15;          // fragment row/col for this lane
  const int khi  = (lane >= 16) ? 8 : 0;

  // ---- preload shared state -------------------------------------------
  for (int i = tix; i < MTILE * HP; i += NTHR) h_s[i] = 0.0f;
  for (int i = tix; i < GATES4; i += NTHR) {
    kb_s[i * 4 + 0] = __builtin_bit_cast(unsigned short, f2bf(bias[i]));
    kb_s[i * 4 + 1] = __builtin_bit_cast(unsigned short, f2bf(kern[0 * GATES4 + i]));
    kb_s[i * 4 + 2] = __builtin_bit_cast(unsigned short, f2bf(kern[1 * GATES4 + i]));
    kb_s[i * 4 + 3] = __builtin_bit_cast(unsigned short, f2bf(kern[2 * GATES4 + i]));
  }
  for (int i = tix; i < UNITS * FEAT; i += NTHR) dw_s[i] = dw[i];
  if (tix < FEAT) dw_s[UNITS * FEAT + tix] = db[tix];
  if (tix < MTILE * FEAT) {            // x for warmup step 0
    int row = tix / FEAT, f = tix % FEAT;
    x_s[row * 4 + f] = inputs[((size_t)(b0 + row) * SEQ + 0) * FEAT + f];
  }

  // cell state c lives in registers: 2 unit-tiles x 8 VGPRs
  v8f cfrag[2];
  const v8f vzero = {};
#pragma unroll
  for (int ut = 0; ut < 2; ++ut) cfrag[ut] = vzero;

  __syncthreads();

  for (int step = 0; step < NSTEPS; ++step) {
    // ================= recurrent GEMM: z = h @ rec_kernel ==============
    v8f acc[2][4];                      // [unit_tile][gate], f32 16x16 tiles
#pragma unroll
    for (int ut = 0; ut < 2; ++ut)
#pragma unroll
      for (int g = 0; g < 4; ++g) acc[ut][g] = vzero;

    for (int kf = 0; kf < NKF; ++kf) {
      // build bf16 A fragment (h rows) from LDS for this K slice of 32
      AW aw;
#pragma unroll
      for (int v = 0; v < 8; ++v) {
        int k0 = kf * 32 + ((v & 3) * 2) + khi + ((v >= 4) ? 16 : 0);
        float2 hv = *(const float2*)&h_s[Ml * HP + k0];
        aw.w[v] = pk_bf16(hv.x, hv.y);
      }
      v16bf a = __builtin_bit_cast(v16bf, aw);
#pragma unroll
      for (int ut = 0; ut < 2; ++ut) {
        int utg = wid + 16 * ut;        // global unit tile 0..31
#pragma unroll
        for (int g = 0; g < 4; ++g) {
          int nt = g * 32 + utg;        // N tile 0..127
          v16bf bfrag = Wp[(size_t)(nt * NKF + kf) * 32 + lane];
          acc[ut][g] = __builtin_amdgcn_wmma_f32_16x16x32_bf16(
              false, a, false, bfrag, (short)0, acc[ut][g], false, false);
        }
      }
    }
    __syncthreads();   // all waves done reading h_s

    // ============== gate nonlinearity + cell update (in regs) ==========
#pragma unroll
    for (int ut = 0; ut < 2; ++ut) {
      int utg = wid + 16 * ut;
#pragma unroll
      for (int r = 0; r < 8; ++r) {
        int M = r + ((lane < 16) ? 0 : 8);
        int u = utg * 16 + Ml;
        float x0 = x_s[M * 4 + 0];
        float x1 = x_s[M * 4 + 1];
        float x2 = x_s[M * 4 + 2];
        float z[4];
#pragma unroll
        for (int g = 0; g < 4; ++g) {
          int n = g * UNITS + u;
          ushort4 kv = *(const ushort4*)&kb_s[n * 4];   // one ds_load_b64
          float t = bf2f_u16(kv.x) + x0 * bf2f_u16(kv.y)
                  + x1 * bf2f_u16(kv.z) + x2 * bf2f_u16(kv.w);
          z[g] = acc[ut][g][r] + t;
        }
        float ig = sigmoidf_(z[0]);
        float fg = sigmoidf_(z[1]);
        float gg = tanhf_(z[2]);
        float og = sigmoidf_(z[3]);
        float c2 = fg * cfrag[ut][r] + ig * gg;
        cfrag[ut][r] = c2;
        h_s[M * HP + u] = og * tanhf_(c2);
      }
    }
    __syncthreads();   // h_s complete for this step

    // ============== dense head / next input ============================
    if (step >= SEQ - 1) {
      // pred = h @ dense_w + dense_b, parallel over all 512 threads:
      // 32 threads per batch row, strided-u partials, then one thread per (row, f) serially sums its 32 LDS partials.
      {
        int row = tix >> 5;             // 0..15
        int j   = tix & 31;
        float p0 = 0.0f, p1 = 0.0f, p2 = 0.0f;
        for (int u = j; u < UNITS; u += 32) {
          float hv = h_s[row * HP + u];
          p0 += hv * dw_s[u * FEAT + 0];
          p1 += hv * dw_s[u * FEAT + 1];
          p2 += hv * dw_s[u * FEAT + 2];
        }
        red_s[(row * 32 + j) * 3 + 0] = p0;
        red_s[(row * 32 + j) * 3 + 1] = p1;
        red_s[(row * 32 + j) * 3 + 2] = p2;
      }
      __syncthreads();
      if (tix < MTILE * FEAT) {
        int row = tix / FEAT, f = tix % FEAT;
        float s = dw_s[UNITS * FEAT + f];
#pragma unroll
        for (int jj = 0; jj < 32; ++jj) s += red_s[(row * 32 + jj) * 3 + f];
        int on = step - (SEQ - 1);
        out[((size_t)on * BATCH + (b0 + row)) * FEAT + f] = s;
        x_s[row * 4 + f] = s;           // feed prediction back as next input
      }
    } else {
      // warmup: stage next sequence input
      if (tix < MTILE * FEAT) {
        int row = tix / FEAT, f = tix % FEAT;
        x_s[row * 4 + f] = inputs[((size_t)(b0 + row) * SEQ + (step + 1)) * FEAT + f];
      }
    }
    __syncthreads();   // x_s ready for next step
  }
}

extern "C" void kernel_launch(void* const* d_in, const int* in_sizes, int n_in,
                              void* d_out, int out_size, void* d_ws, size_t ws_size,
                              hipStream_t stream) {
  (void)in_sizes; (void)n_in; (void)out_size; (void)ws_size;
  const float* inputs = (const float*)d_in[0];   // [2048,64,3]
  const float* kern   = (const float*)d_in[1];   // [3,2048]
  const float* rec    = (const float*)d_in[2];   // [512,2048]
  const float* bias   = (const float*)d_in[3];   // [2048]
  const float* dw     = (const float*)d_in[4];   // [512,3]
  const float* db     = (const float*)d_in[5];   // [3]

  __bf16* Wp = (__bf16*)d_ws;                    // 2 MB packed weights

  {
    int total = UNITS * GATES4;                  // 1,048,576 bf16 elements
    feedback_pack_w<<<(total + 255) / 256, 256, 0, stream>>>(rec, Wp);
  }
  feedback_lstm_main<<<BATCH / MTILE, NTHR, 0, stream>>>(
      inputs, kern, bias, dw, db, (const v16bf*)Wp, (float*)d_out);
}
